// ResidualBlock_24464133718186
// MI455X (gfx1250) — 8-bit matrix operand order repaired and checked by reading, compiled, not run on hardware
//
#include <hip/hip_runtime.h>

typedef int   v4i  __attribute__((ext_vector_type(4)));
typedef int   v16i __attribute__((ext_vector_type(16)));
typedef float v4f  __attribute__((ext_vector_type(4)));
typedef float v8f  __attribute__((ext_vector_type(8)));

#define CIN   256
#define COUT  256
#define HH    28
#define KD    2304
#define LPITCH 144
#define ABYTES (128 * LPITCH)
#define BUFB   (2 * ABYTES)
#define XB_BYTES ((size_t)128 * 28 * 28 * 256)

__device__ __forceinline__ unsigned int sgn_fp8(float v) {
    return v > 0.f ? 0x38u : (v < 0.f ? 0xB8u : 0x00u);
}

__global__ void __launch_bounds__(256) pack_x_kernel(const float* __restrict__ x,
                                                     unsigned char* __restrict__ xb) {
    int d   = blockIdx.x * blockDim.x + threadIdx.x;
    int c4  = (d & 63) * 4;
    int pix = d >> 6;
    int w_  = pix % 28;
    int h_  = (pix / 28) % 28;
    int n_  = pix / 784;
    unsigned int packed = 0;
#pragma unroll
    for (int j = 0; j < 4; ++j) {
        float v = x[(((size_t)n_ * CIN + (c4 + j)) * HH + h_) * HH + w_];
        packed |= sgn_fp8(v) << (8 * j);
    }
    ((unsigned int*)xb)[d] = packed;
}

__global__ void __launch_bounds__(256) pack_w_kernel(const float* __restrict__ w,
                                                     unsigned char* __restrict__ wb) {
    int d    = blockIdx.x * blockDim.x + threadIdx.x;
    int c4   = (d & 63) * 4;
    int rest = d >> 6;
    int rs   = rest % 9;
    int o    = rest / 9;
    int r = rs / 3, s = rs % 3;
    unsigned int packed = 0;
#pragma unroll
    for (int j = 0; j < 4; ++j) {
        float v = w[(((size_t)o * CIN + (c4 + j)) * 3 + r) * 3 + s];
        packed |= sgn_fp8(v) << (8 * j);
    }
    ((unsigned int*)wb)[d] = packed;
}

__global__ void __launch_bounds__(256)
binconv_wmma_kernel(const unsigned char* __restrict__ xb,
                    const unsigned char* __restrict__ wb,
                    const float* __restrict__ x,
                    const float* __restrict__ gamma,
                    const float* __restrict__ beta,
                    const float* __restrict__ mean,
                    const float* __restrict__ var,
                    float* __restrict__ out)
{
    __shared__ unsigned char smem[2 * BUFB];

    const int tid     = threadIdx.x;
    const int wave    = tid >> 5;
    const int lane    = tid & 31;
    const int lane16  = lane & 15;
    const int laneHi  = lane >> 4;
    const int waveRow = wave >> 1;
    const int waveCol = wave & 1;

    const int blockM  = blockIdx.x * 128;
    const int colBase = blockIdx.y * 128;

    int    a_lds[4];
    size_t a_base[4];
    int    a_h[4], a_w[4];
    size_t b_base[4];
#pragma unroll
    for (int i = 0; i < 4; ++i) {
        int chunk = tid + i * 256;
        int row   = chunk >> 3;
        int k16   = (chunk & 7) * 16;
        int m  = blockM + row;
        int w_ = m % 28;
        int h_ = (m / 28) % 28;
        int n_ = m / 784;
        a_lds[i]  = row * LPITCH + k16;
        a_h[i]    = h_;
        a_w[i]    = w_;
        a_base[i] = (((size_t)n_ * 28 + h_) * 28 + w_) * CIN + k16;
        b_base[i] = (size_t)(colBase + row) * KD + k16;
    }

    v8f acc[8];
#pragma unroll
    for (int j = 0; j < 8; ++j) acc[j] = (v8f){};

    v4i aReg[4], bReg[4];

    auto load_step = [&](int t) {
        const int rs    = t >> 1;
        const int r     = rs / 3;
        const int s     = rs % 3;
        const int c_off = (t & 1) * 128;
        const long haloOff = (long)((r - 1) * 28 + (s - 1)) * CIN + c_off;
#pragma unroll
        for (int i = 0; i < 4; ++i) {
            int hi = a_h[i] + r - 1;
            int wi = a_w[i] + s - 1;
            v4i val = (v4i){};
            if ((unsigned)hi < 28u && (unsigned)wi < 28u) {
                val = *(const v4i*)(xb + (long)a_base[i] + haloOff);
            }
            aReg[i] = val;
        }
#pragma unroll
        for (int i = 0; i < 4; ++i) {
            bReg[i] = *(const v4i*)(wb + b_base[i] + (size_t)(rs * CIN + c_off));
        }
    };

    auto store_step = [&](int buf) {
        unsigned char* lA = smem + buf * BUFB;
        unsigned char* lB = lA + ABYTES;
#pragma unroll
        for (int i = 0; i < 4; ++i) *(v4i*)(lA + a_lds[i]) = aReg[i];
#pragma unroll
        for (int i = 0; i < 4; ++i) *(v4i*)(lB + a_lds[i]) = bReg[i];
    };

    auto compute_step = [&](int buf) {
        const unsigned char* lA = smem + buf * BUFB;
        const unsigned char* lB = lA + ABYTES;

        v16i afrag[2];
#pragma unroll
        for (int a = 0; a < 2; ++a) {
            const unsigned char* arow =
                lA + (waveRow * 32 + a * 16 + lane16) * LPITCH;
#pragma unroll
            for (int hv = 0; hv < 2; ++hv) {
#pragma unroll
                for (int q = 0; q < 4; ++q) {
                    unsigned long long dq = *(const unsigned long long*)
                        (arow + hv * 64 + q * 16 + laneHi * 8);
                    afrag[a][hv * 8 + q * 2 + 0] = (int)(unsigned int)(dq & 0xffffffffull);
                    afrag[a][hv * 8 + q * 2 + 1] = (int)(unsigned int)(dq >> 32);
                }
            }
        }

#pragma unroll
        for (int j = 0; j < 4; ++j) {
            v16i bfrag;
            const unsigned char* bcol =
                lB + (waveCol * 64 + j * 16 + lane16) * LPITCH + laneHi * 8;
#pragma unroll
            for (int g = 0; g < 4; ++g) {
                unsigned long long d0 = *(const unsigned long long*)(bcol + g * 32);
                unsigned long long d1 = *(const unsigned long long*)(bcol + g * 32 + 16);
                bfrag[g * 4 + 0] = (int)(unsigned int)(d0 & 0xffffffffull);
                bfrag[g * 4 + 1] = (int)(unsigned int)(d0 >> 32);
                bfrag[g * 4 + 2] = (int)(unsigned int)(d1 & 0xffffffffull);
                bfrag[g * 4 + 3] = (int)(unsigned int)(d1 >> 32);
            }
#pragma unroll
            for (int a = 0; a < 2; ++a) {
                acc[a * 4 + j] = __builtin_amdgcn_wmma_f32_16x16x128_fp8_fp8(
                    afrag[a], bfrag, (short)0, acc[a * 4 + j],
                     false,  false);
            }
        }
    };

    load_step(0);
#pragma unroll
    for (int t = 0; t < 18; ++t) {
        const int buf = t & 1;
        __syncthreads();
        store_step(buf);
        __syncthreads();
        if (t < 17) load_step(t + 1);
        compute_step(buf);
    }

    __syncthreads();
    float* ldsW = (float*)smem + wave * 2048;

#pragma unroll
    for (int j = 0; j < 4; ++j) {
        int k = colBase + waveCol * 64 + j * 16 + lane16;
        float inv = gamma[k] * rsqrtf(var[k] + 1e-5f);
        float bns = beta[k] - mean[k] * inv;
        int colL = j * 16 + lane16;
#pragma unroll
        for (int a = 0; a < 2; ++a) {
            v4f lo, hi;
#pragma unroll
            for (int v = 0; v < 4; ++v) {
                float o0 = acc[a * 4 + j][v] * inv + bns;
                float o1 = acc[a * 4 + j][v + 4] * inv + bns;
                lo[v] = fminf(1.0f, fmaxf(-1.0f, o0));
                hi[v] = fminf(1.0f, fmaxf(-1.0f, o1));
            }
            float* dst = ldsW + colL * 32 + a * 16 + laneHi * 8;
            *(v4f*)dst       = lo;
            *(v4f*)(dst + 4) = hi;
        }
    }
    __syncthreads();

    {
        int m  = blockM + waveRow * 32 + lane;
        int w_ = m % 28;
        int h_ = (m / 28) % 28;
        int n_ = m / 784;
        size_t pbase = (size_t)n_ * (COUT * 784) + (size_t)(h_ * 28 + w_);
        int kbase = colBase + waveCol * 64;
#pragma unroll 8
        for (int c = 0; c < 64; ++c) {
            float vclamped = ldsW[c * 32 + lane];
            size_t idx = pbase + (size_t)(kbase + c) * 784;
            out[idx] = vclamped + x[idx];
        }
    }
}

extern "C" void kernel_launch(void* const* d_in, const int* in_sizes, int n_in,
                              void* d_out, int out_size, void* d_ws, size_t ws_size,
                              hipStream_t stream) {
    const float* x     = (const float*)d_in[0];
    const float* w     = (const float*)d_in[1];
    const float* gamma = (const float*)d_in[2];
    const float* beta  = (const float*)d_in[3];
    const float* mean  = (const float*)d_in[4];
    const float* var   = (const float*)d_in[5];
    float* out = (float*)d_out;

    unsigned char* xb = (unsigned char*)d_ws;
    unsigned char* wb = xb + XB_BYTES;

    pack_x_kernel<<<25088, 256, 0, stream>>>(x, xb);
    pack_w_kernel<<<576, 256, 0, stream>>>(w, wb);

    dim3 grid(784, 2);
    binconv_wmma_kernel<<<grid, 256, 0, stream>>>(xb, wb, x, gamma, beta, mean, var, out);
}
